// ActorCritic_28415503631082
// MI455X (gfx1250) — matrix operand order repaired and checked by reading, not compiled, not run on hardware
//
#include <hip/hip_runtime.h>
#include <cstdint>
#include <cstddef>

// ---------------------------------------------------------------------------
// Types
// ---------------------------------------------------------------------------
typedef __attribute__((ext_vector_type(16))) __bf16 v16bf;
typedef __attribute__((ext_vector_type(8)))  __bf16 v8bf;
typedef __attribute__((ext_vector_type(8)))  float  v8f;
typedef __attribute__((ext_vector_type(4)))  float  v4f;

// Problem constants (match reference)
#define JJ    16
#define MM    4096
#define FF    2048
#define AA    32
#define HIDN  400
#define NTOT  25          // 400 / 16
#define KT_H  13          // ceil(400/32) -> K padded to 416
#define PITCH 416
#define ROWS_ALL 65536    // J*M
#define ROWS_ACT 61440    // (J-1)*M
#define TJ 15             // J-1

__device__ __forceinline__ __bf16 cvt_bf16(float f) {
  uint32_t u = __builtin_bit_cast(uint32_t, f);
  u += 0x7FFFu + ((u >> 16) & 1u);            // round to nearest even
  unsigned short h = (unsigned short)(u >> 16);
  return __builtin_bit_cast(__bf16, h);
}

__device__ __forceinline__ __bf16 bf16_zero() {
  return __builtin_bit_cast(__bf16, (unsigned short)0);
}

// ---------------------------------------------------------------------------
// Weight packing: f32 row-major W[K][N] -> bf16 WMMA B-fragments.
// B (32x16, bf16) register layout: lane L holds column n = L%16,
// K values (L/16)*8 .. +7 then 16+(L/16)*8 .. +7, two per VGPR. We store each lane's 16
// K values contiguously (32B) so the GEMM loads them as b128 pairs.
// frag index = ((kt*NT + nt)*32 + lane)*16
// ---------------------------------------------------------------------------
__global__ __launch_bounds__(256) void ac_pack_w(
    const float* __restrict__ W, int K, int N, int KT, int NT,
    __bf16* __restrict__ out)
{
  int gid = blockIdx.x * blockDim.x + threadIdx.x;
  int total = KT * NT * 32;
  if (gid >= total) return;
  int lane = gid & 31;
  int nt   = (gid >> 5) % NT;
  int kt   = (gid >> 5) / NT;
  int n     = nt * 16 + (lane & 15);
  int kbase = kt * 32 + (lane >> 4) * 8;
  __bf16* dst = out + (size_t)gid * 16;
#pragma unroll
  for (int j = 0; j < 16; ++j) {
    int k = kbase + (j < 8 ? j : j + 8);
    float v = (k < K && n < N) ? W[(size_t)k * N + n] : 0.0f;
    dst[j] = cvt_bf16(v);
  }
}

// ---------------------------------------------------------------------------
// Fused hidden layer: OUT = ELU(LN(IN @ W + b)) ; OUT is bf16, pitch 416
// (cols 400..415 zero). Block = 256 thr = 8 waves, 32 rows, full N=400.
// Waves: 2 (M) x 4 (N); each wave 6-7 N-tiles of 16.
// ---------------------------------------------------------------------------
template <bool IN_F32>
__global__ __launch_bounds__(256) void ac_mlp_hidden(
    const void* __restrict__ in, int in_pitch, int KT,
    const __bf16* __restrict__ wfrag,
    const float* __restrict__ bias,
    const float* __restrict__ ln_g,
    const float* __restrict__ ln_b,
    __bf16* __restrict__ out)
{
  __shared__ float smem[32][404];
  __shared__ float psum[32][8];
  __shared__ float psq[32][8];
  __shared__ float s_mu[32];
  __shared__ float s_rs[32];

  const int tid   = threadIdx.x;
  const int lane  = tid & 31;
  const int wave  = tid >> 5;
  const int mtile = wave & 1;
  const int ngrp  = wave >> 1;
  const int ntstart = (ngrp == 0) ? 0 : (1 + ngrp * 6);   // 0,7,13,19
  const int ncount  = (ngrp == 0) ? 7 : 6;                // covers 0..24
  const size_t row0 = (size_t)blockIdx.x * 32;

  const int khalf = lane >> 4;                 // ISA A-layout half select
  const size_t grow = row0 + (size_t)mtile * 16 + (size_t)(lane & 15);

  v8f acc[7];
#pragma unroll
  for (int t = 0; t < 7; ++t)
#pragma unroll
    for (int r = 0; r < 8; ++r) acc[t][r] = 0.0f;

  for (int kt = 0; kt < KT; ++kt) {
    if (kt + 1 < KT) {   // prefetch next K-tile's B fragments (L2 resident)
      __builtin_prefetch(
          (const void*)(wfrag + ((size_t)((kt + 1) * NTOT + ntstart) * 32 + lane) * 16),
          0, 1);
    }

    // --- load A fragment (16x32 bf16 ISA layout): two 8-elem K runs/lane ---
    v16bf a;
    if (IN_F32) {
      const float* ip = (const float*)in + grow * (size_t)in_pitch
                        + (size_t)kt * 32 + (size_t)khalf * 8;
      v4f x0 = *(const v4f*)(ip + 0);
      v4f x1 = *(const v4f*)(ip + 4);
      v4f x2 = *(const v4f*)(ip + 16);
      v4f x3 = *(const v4f*)(ip + 20);
#pragma unroll
      for (int j = 0; j < 4; ++j) {
        a[j]      = cvt_bf16(x0[j]);
        a[4 + j]  = cvt_bf16(x1[j]);
        a[8 + j]  = cvt_bf16(x2[j]);
        a[12 + j] = cvt_bf16(x3[j]);
      }
    } else {
      const __bf16* ip = (const __bf16*)in + grow * (size_t)in_pitch
                         + (size_t)kt * 32 + (size_t)khalf * 8;
      v8bf lo = *(const v8bf*)(ip);
      v8bf hi = *(const v8bf*)(ip + 16);
#pragma unroll
      for (int j = 0; j < 8; ++j) { a[j] = lo[j]; a[8 + j] = hi[j]; }
    }

    // --- MAC over this wave's N tiles ---
#pragma unroll
    for (int t = 0; t < 7; ++t) {
      if (t < ncount) {
        const int nt = ntstart + t;
        const v16bf b =
            ((const v16bf*)wfrag)[(size_t)(kt * NTOT + nt) * 32 + lane];
        acc[t] = __builtin_amdgcn_wmma_f32_16x16x32_bf16(
            false, a, false, b, (short)0, acc[t], false, false);
      }
    }
  }

  // --- C tiles -> LDS (C layout: vgpr r -> row r + 8*(lane/16)) ---
  {
    const int nc0   = lane & 15;
    const int rbase = mtile * 16 + khalf * 8;
#pragma unroll
    for (int t = 0; t < 7; ++t) {
      if (t < ncount) {
        const int nc = (ntstart + t) * 16 + nc0;
#pragma unroll
        for (int r = 0; r < 8; ++r) smem[rbase + r][nc] = acc[t][r];
      }
    }
  }
  __syncthreads();

  // --- bias + LayerNorm + ELU, 8 threads per row ---
  const int lrow = tid >> 3;
  const int sub  = tid & 7;
  {
    float s = 0.f, s2 = 0.f;
    const int c0 = sub * 50;
    for (int c = c0; c < c0 + 50; ++c) {
      float x = smem[lrow][c] + bias[c];
      s += x; s2 += x * x;
    }
    psum[lrow][sub] = s; psq[lrow][sub] = s2;
  }
  __syncthreads();
  if (tid < 32) {
    float s = 0.f, s2 = 0.f;
#pragma unroll
    for (int j = 0; j < 8; ++j) { s += psum[tid][j]; s2 += psq[tid][j]; }
    const float mu  = s * (1.0f / 400.0f);
    const float var = s2 * (1.0f / 400.0f) - mu * mu;
    s_mu[tid] = mu;
    s_rs[tid] = rsqrtf(var + 1e-5f);
  }
  __syncthreads();
  {
    const float mu = s_mu[lrow];
    const float rs = s_rs[lrow];
    __bf16* orow = out + (row0 + (size_t)lrow) * PITCH;
    const int c0 = sub * 50;
    for (int c = c0; c < c0 + 50; ++c) {
      float x = smem[lrow][c] + bias[c];
      float y = (x - mu) * rs * ln_g[c] + ln_b[c];
      float e = (y > 0.f) ? y : (__expf(y) - 1.f);
      orow[c] = cvt_bf16(e);
    }
    const int p0 = 400 + sub * 2;        // zero the K padding
    orow[p0]     = bf16_zero();
    orow[p0 + 1] = bf16_zero();
  }
}

// ---------------------------------------------------------------------------
// Head: OUT = IN @ Wh + bh (no LN/ELU). NT=2,Ncols=32 (actor) writes row-major
// logits; NT=1,Ncols=1 (critic) writes a value vector.
// ---------------------------------------------------------------------------
__global__ __launch_bounds__(256) void ac_head(
    const __bf16* __restrict__ in,      // pitch 416
    const __bf16* __restrict__ wfrag,   // KT=13, NT tiles
    const float* __restrict__ bias,
    float* __restrict__ outF,
    int NT, int Ncols)
{
  const int tid   = threadIdx.x;
  const int lane  = tid & 31;
  const int wave  = tid >> 5;
  const int ntile = wave % NT;
  const int mtile = wave / NT;
  const int mblock = (8 / NT) * 16;
  const size_t row0 = (size_t)blockIdx.x * (size_t)mblock;
  const int khalf = lane >> 4;
  const size_t grow = row0 + (size_t)mtile * 16 + (size_t)(lane & 15);

  v8f acc;
#pragma unroll
  for (int r = 0; r < 8; ++r) acc[r] = 0.0f;

  for (int kt = 0; kt < KT_H; ++kt) {
    const __bf16* ip = in + grow * (size_t)PITCH + (size_t)kt * 32 + (size_t)khalf * 8;
    v8bf lo = *(const v8bf*)(ip);
    v8bf hi = *(const v8bf*)(ip + 16);
    v16bf a;
#pragma unroll
    for (int j = 0; j < 8; ++j) { a[j] = lo[j]; a[8 + j] = hi[j]; }
    const v16bf b = ((const v16bf*)wfrag)[(size_t)(kt * NT + ntile) * 32 + lane];
    acc = __builtin_amdgcn_wmma_f32_16x16x32_bf16(
        false, a, false, b, (short)0, acc, false, false);
  }

  const int col   = ntile * 16 + (lane & 15);
  const int rbase = mtile * 16 + khalf * 8;
  if (Ncols == 1) {
    if ((lane & 15) == 0) {
      const float bb = bias[0];
#pragma unroll
      for (int r = 0; r < 8; ++r)
        outF[row0 + (size_t)(rbase + r)] = acc[r] + bb;
    }
  } else {
    if (col < Ncols) {
      const float bb = bias[col];
#pragma unroll
      for (int r = 0; r < 8; ++r)
        outF[(row0 + (size_t)(rbase + r)) * (size_t)Ncols + col] = acc[r] + bb;
    }
  }
}

// ---------------------------------------------------------------------------
// advantage[j,m] = -v_t[j,m] + r[j+1,m] + GAMMA*(1-t[j+1,m])*v_t[j+1,m]
// ---------------------------------------------------------------------------
__global__ __launch_bounds__(256) void ac_adv(
    const float* __restrict__ value_t, const float* __restrict__ rewards,
    const float* __restrict__ terminals, float* __restrict__ adv)
{
  int i = blockIdx.x * blockDim.x + threadIdx.x;
  if (i >= TJ * MM) return;
  float v0t = value_t[i];
  float v1t = value_t[i + MM];
  float r1  = rewards[i + MM];
  float t1  = terminals[i + MM];
  adv[i] = -v0t + r1 + 0.999f * (1.f - t1) * v1t;
}

// ---------------------------------------------------------------------------
// Reverse GAE scan over j (15 steps) per m; also value_target and reality wt.
// ---------------------------------------------------------------------------
__global__ __launch_bounds__(256) void ac_gae(
    const float* __restrict__ adv, const float* __restrict__ terminals,
    const float* __restrict__ value_t,
    float* __restrict__ gae, float* __restrict__ vtgt, float* __restrict__ rw)
{
  int m = blockIdx.x * blockDim.x + threadIdx.x;
  if (m >= MM) return;
  float c = 0.f;
  for (int j = TJ - 1; j >= 0; --j) {
    float t1 = terminals[(j + 1) * MM + m];
    c = adv[j * MM + m] + (0.95f * 0.999f) * (1.f - t1) * c;
    gae[j * MM + m]  = c;
    vtgt[j * MM + m] = c + value_t[j * MM + m];
  }
  float p = 1.f;
  for (int j = 0; j < TJ; ++j) {
    p *= (1.f - terminals[j * MM + m]);
    rw[j * MM + m] = p;
  }
}

__device__ __forceinline__ float ac_block_sum(float v, float* sdata) {
  int t = threadIdx.x;
  sdata[t] = v;
  __syncthreads();
  for (int s = 128; s > 0; s >>= 1) {
    if (t < s) sdata[t] += sdata[t + s];
    __syncthreads();
  }
  return sdata[0];
}

__global__ __launch_bounds__(256) void ac_critic_loss(
    const float* __restrict__ vtgt, const float* __restrict__ value0,
    const float* __restrict__ rw, float* __restrict__ partials)
{
  __shared__ float sdata[256];
  int i = blockIdx.x * blockDim.x + threadIdx.x;
  float v = 0.f;
  if (i < TJ * MM) {
    float e = vtgt[i] - value0[i];
    v = 0.5f * e * e * rw[i];
  }
  float s = ac_block_sum(v, sdata);
  if (threadIdx.x == 0) partials[blockIdx.x] = s;
}

__global__ __launch_bounds__(256) void ac_actor_loss(
    const float* __restrict__ logits, const int* __restrict__ actions,
    const float* __restrict__ gae, const float* __restrict__ rw,
    float* __restrict__ partials)
{
  __shared__ float sdata[256];
  int i = blockIdx.x * blockDim.x + threadIdx.x;
  float v = 0.f;
  if (i < TJ * MM) {
    const float* l = logits + (size_t)i * AA;
    float mx = l[0];
    for (int k = 1; k < AA; ++k) mx = fmaxf(mx, l[k]);
    float se = 0.f;
    for (int k = 0; k < AA; ++k) se += __expf(l[k] - mx);
    float lse = mx + __logf(se);
    float inv = 1.f / se;
    float pl = 0.f;
    for (int k = 0; k < AA; ++k) pl += __expf(l[k] - mx) * inv * l[k];
    float entropy = lse - pl;
    int a = actions[i];
    float lpa = l[a] - lse;
    v = (-lpa * gae[i] - 0.001f * entropy) * rw[i];
  }
  float s = ac_block_sum(v, sdata);
  if (threadIdx.x == 0) partials[blockIdx.x] = s;
}

__global__ __launch_bounds__(256) void ac_finalize(
    const float* __restrict__ pa, const float* __restrict__ pc, int n,
    float* __restrict__ out)
{
  __shared__ float sa[256];
  __shared__ float sc[256];
  int t = threadIdx.x;
  sa[t] = (t < n) ? pa[t] : 0.f;
  sc[t] = (t < n) ? pc[t] : 0.f;
  __syncthreads();
  for (int s = 128; s > 0; s >>= 1) {
    if (t < s) { sa[t] += sa[t + s]; sc[t] += sc[t + s]; }
    __syncthreads();
  }
  if (t == 0) {
    out[0] = sa[0] * (1.0f / (float)(TJ * MM));
    out[1] = sc[0] * (1.0f / (float)(TJ * MM));
  }
}

// ---------------------------------------------------------------------------
// Workspace layout
// ---------------------------------------------------------------------------
static constexpr size_t FRAG0_SZ = (size_t)64 * 25 * 32 * 16 * 2;   // 1,638,400
static constexpr size_t FRAGL_SZ = (size_t)13 * 25 * 32 * 16 * 2;   //   332,800
static constexpr size_t FRAGH_SZ = (size_t)13 * 2 * 32 * 16 * 2;    //    26,624
static constexpr size_t ACT_SZ   = (size_t)ROWS_ALL * PITCH * 2;    // 54,525,952
static constexpr size_t VAL_SZ   = (size_t)ROWS_ALL * 4;
static constexpr size_t LOG_SZ   = (size_t)ROWS_ACT * AA * 4;
static constexpr size_t SM_SZ    = (size_t)TJ * MM * 4;

static constexpr size_t OFF_FRAG0  = 0;
static constexpr size_t OFF_FRAG1  = OFF_FRAG0 + FRAG0_SZ;
static constexpr size_t OFF_FRAG2  = OFF_FRAG1 + FRAGL_SZ;
static constexpr size_t OFF_FRAG3  = OFF_FRAG2 + FRAGL_SZ;
static constexpr size_t OFF_FRAGH  = OFF_FRAG3 + FRAGL_SZ;
static constexpr size_t OFF_ACTA   = OFF_FRAGH + FRAGH_SZ;
static constexpr size_t OFF_ACTB   = OFF_ACTA + ACT_SZ;
static constexpr size_t OFF_VALT   = OFF_ACTB + ACT_SZ;
static constexpr size_t OFF_VAL0   = OFF_VALT + VAL_SZ;
static constexpr size_t OFF_LOGITS = OFF_VAL0 + VAL_SZ;
static constexpr size_t OFF_ADV    = OFF_LOGITS + LOG_SZ;
static constexpr size_t OFF_GAE    = OFF_ADV + SM_SZ;
static constexpr size_t OFF_VTGT   = OFF_GAE + SM_SZ;
static constexpr size_t OFF_RW     = OFF_VTGT + SM_SZ;
static constexpr size_t OFF_PA     = OFF_RW + SM_SZ;
static constexpr size_t OFF_PC     = OFF_PA + 1024;

// ---------------------------------------------------------------------------
// Host launcher
// d_in: [0]=features [1]=rewards [2]=terminals
//       [3..20]=actor params, [21..38]=critic params, [39..56]=target params,
//       [57]=actions_idx   (each param list: (W,b,g,be)*4, Wh, bh)
// ---------------------------------------------------------------------------
extern "C" void kernel_launch(void* const* d_in, const int* in_sizes, int n_in,
                              void* d_out, int out_size, void* d_ws, size_t ws_size,
                              hipStream_t stream) {
  (void)in_sizes; (void)n_in; (void)out_size; (void)ws_size;

  const float* feats     = (const float*)d_in[0];
  const float* rewards   = (const float*)d_in[1];
  const float* terminals = (const float*)d_in[2];
  const int*   actions   = (const int*)d_in[57];

  uint8_t* ws = (uint8_t*)d_ws;
  __bf16* frag0 = (__bf16*)(ws + OFF_FRAG0);
  __bf16* fragL[3] = { (__bf16*)(ws + OFF_FRAG1), (__bf16*)(ws + OFF_FRAG2),
                       (__bf16*)(ws + OFF_FRAG3) };
  __bf16* fragH = (__bf16*)(ws + OFF_FRAGH);
  __bf16* actA  = (__bf16*)(ws + OFF_ACTA);
  __bf16* actB  = (__bf16*)(ws + OFF_ACTB);
  float* value_t = (float*)(ws + OFF_VALT);
  float* value0  = (float*)(ws + OFF_VAL0);
  float* logits  = (float*)(ws + OFF_LOGITS);
  float* adv  = (float*)(ws + OFF_ADV);
  float* gae  = (float*)(ws + OFF_GAE);
  float* vtgt = (float*)(ws + OFF_VTGT);
  float* rw   = (float*)(ws + OFF_RW);
  float* pa   = (float*)(ws + OFF_PA);
  float* pc   = (float*)(ws + OFF_PC);

  // Process: critic_target -> value_t ; critic -> value0 ; actor -> logits
  const int bases[3] = { 39, 21, 3 };
  for (int mi = 0; mi < 3; ++mi) {
    const int base = bases[mi];
    const bool isActor = (base == 3);
    const int rows = isActor ? ROWS_ACT : ROWS_ALL;
    const int headN  = isActor ? AA : 1;
    const int headNT = isActor ? 2 : 1;

    // Pack weights into WMMA B-fragment layout (bf16)
    ac_pack_w<<<(64 * 25 * 32 + 255) / 256, 256, 0, stream>>>(
        (const float*)d_in[base + 0], FF, HIDN, 64, NTOT, frag0);
    for (int l = 1; l < 4; ++l)
      ac_pack_w<<<(13 * 25 * 32 + 255) / 256, 256, 0, stream>>>(
          (const float*)d_in[base + 4 * l], HIDN, HIDN, KT_H, NTOT, fragL[l - 1]);
    ac_pack_w<<<(13 * headNT * 32 + 255) / 256, 256, 0, stream>>>(
        (const float*)d_in[base + 16], HIDN, headN, KT_H, headNT, fragH);

    // Layer 0 (f32 features in, K=2048)
    ac_mlp_hidden<true><<<rows / 32, 256, 0, stream>>>(
        (const void*)feats, FF, 64, frag0,
        (const float*)d_in[base + 1], (const float*)d_in[base + 2],
        (const float*)d_in[base + 3], actA);
    // Layers 1..3 (bf16 in, K=416)
    __bf16* bufs[2] = { actA, actB };
    for (int l = 1; l < 4; ++l) {
      ac_mlp_hidden<false><<<rows / 32, 256, 0, stream>>>(
          (const void*)bufs[(l + 1) & 1], PITCH, KT_H, fragL[l - 1],
          (const float*)d_in[base + 4 * l + 1], (const float*)d_in[base + 4 * l + 2],
          (const float*)d_in[base + 4 * l + 3], bufs[l & 1]);
    }
    // Head (input is bufs[1] == actB after 4 layers)
    float* hout = isActor ? logits : ((mi == 0) ? value_t : value0);
    const int mblock = (8 / headNT) * 16;   // 64 (actor) / 128 (critic)
    ac_head<<<rows / mblock, 256, 0, stream>>>(
        actB, fragH, (const float*)d_in[base + 17], hout, headNT, headN);
  }

  // GAE + losses
  ac_adv<<<(TJ * MM + 255) / 256, 256, 0, stream>>>(value_t, rewards, terminals, adv);
  ac_gae<<<(MM + 255) / 256, 256, 0, stream>>>(adv, terminals, value_t, gae, vtgt, rw);

  const int nred = (TJ * MM + 255) / 256;   // 240
  ac_critic_loss<<<nred, 256, 0, stream>>>(vtgt, value0, rw, pc);
  ac_actor_loss<<<nred, 256, 0, stream>>>(logits, actions, gae, rw, pa);
  ac_finalize<<<1, 256, 0, stream>>>(pa, pc, nred, (float*)d_out);
}
